// Single_Check_70789650973099
// MI455X (gfx1250) — matrix operand order repaired and checked by reading, compiled, not run on hardware
//
#include <hip/hip_runtime.h>

typedef __attribute__((ext_vector_type(16))) _Float16 v16h;
typedef __attribute__((ext_vector_type(8)))  float    v8f;
typedef __attribute__((ext_vector_type(4)))  float    v4f;

constexpr int cB    = 256;        // batch
constexpr int cS    = 128;        // sequence length
constexpr int cW    = 6;          // pad / window-1
constexpr int cT    = cW + 1;     // 7 timesteps
constexpr int cE    = 128;        // embedding dim
constexpr int cH    = 128;        // hidden dim
constexpr int cG    = 4 * cH;     // 512 gates per direction
constexpr int cNG   = 2 * cG;     // 1024 gate columns (fwd|bwd)
constexpr int cPOS  = cS + cW;    // 134 used positions per batch
constexpr int cPSTR = 144;        // padded positions per batch (9 tiles of 16)

// P layout (gate-major): P[((b*cNG + n) * cPSTR) + p]  (f16)

// ---------------- fast math on the recurrence critical path ----------------
__device__ __forceinline__ float frcp(float x) { return __builtin_amdgcn_rcpf(x); }
__device__ __forceinline__ float sigm(float x) { return frcp(1.f + __expf(-x)); }
__device__ __forceinline__ float tanh_fast(float x) {
  float e = __expf(-2.f * x);
  return (1.f - e) * frcp(1.f + e);
}

// ---------------- WMMA helpers (gfx1250 wave32 layouts) ----------------

__device__ __forceinline__ v8f wmma_f16(v16h a, v16h b, v8f c) {
  return __builtin_amdgcn_wmma_f32_16x16x32_f16(false, a, false, b, (short)0, c,
                                                false, false);
}

// A fragment 16x32 f16 from a [16][128] row-major f16 LDS tile.
__device__ __forceinline__ v16h load_a_lds(const _Float16* base, int lane, int kk) {
  int m  = lane & 15;
  int k0 = kk * 32 + ((lane >> 4) & 1) * 8;
  const _Float16* p = base + m * 128 + k0;
  union { v16h v; uint4 q[2]; } u;
  u.q[0] = *reinterpret_cast<const uint4*>(p);
  u.q[1] = *reinterpret_cast<const uint4*>(p + 16);
  return u.v;
}

// B fragment 32x16 f16 from row-major f16 weight [N][ldk]; B[k][n] = Wt[n][k].
__device__ __forceinline__ v16h load_b_h(const _Float16* Wt, int n0, int ldk,
                                         int lane, int kk) {
  int n  = n0 + (lane & 15);
  int k0 = kk * 32 + ((lane >> 4) & 1) * 8;
  const _Float16* p = Wt + (size_t)n * ldk + k0;
  union { v16h v; uint4 q[2]; } u;
  u.q[0] = *reinterpret_cast<const uint4*>(p);
  u.q[1] = *reinterpret_cast<const uint4*>(p + 16);
  return u.v;
}

// A fragment for FC1: K<128 from hf (saved forward h), K>=128 from hb (current
// backward h); leaky_relu(0.01) applied in f16.
__device__ __forceinline__ v16h load_a_fc(const _Float16* hf, const _Float16* hb,
                                          int lane, int kk) {
  int m  = lane & 15;
  int k0 = kk * 32 + ((lane >> 4) & 1) * 8;
  union { v16h v; uint4 q[2]; _Float16 h[16]; } u;
#pragma unroll
  for (int half = 0; half < 2; ++half) {
    int kb = k0 + half * 16;
    const _Float16* src = (kb < cH) ? (hf + m * 128 + kb)
                                    : (hb + m * 128 + (kb - cH));
    u.q[half] = *reinterpret_cast<const uint4*>(src);
  }
#pragma unroll
  for (int i = 0; i < 16; ++i) {
    _Float16 x = u.h[i];
    u.h[i] = x > (_Float16)0.f ? x : x * (_Float16)0.01f;
  }
  return u.v;
}

// ---------------- Kernel 0: one-time weight convert / fuse ----------------
__global__ __launch_bounds__(256) void k_cvt(
    const float* __restrict__ w_ih_f, const float* __restrict__ w_ih_b,
    const float* __restrict__ w_hh_f, const float* __restrict__ w_hh_b,
    const float* __restrict__ w1,
    const float* __restrict__ b_ih_f, const float* __restrict__ b_hh_f,
    const float* __restrict__ b_ih_b, const float* __restrict__ b_hh_b,
    _Float16* __restrict__ Wih, _Float16* __restrict__ Whh,
    _Float16* __restrict__ W1h, float* __restrict__ biasC) {
  const int i = blockIdx.x * 256 + threadIdx.x;
  const int WQ = cG * cE;            // 65536 per direction
  if (i < WQ) {
    Wih[i] = (_Float16)w_ih_f[i];
    Whh[i] = (_Float16)w_hh_f[i];
    Wih[WQ + i] = (_Float16)w_ih_b[i];
    Whh[WQ + i] = (_Float16)w_hh_b[i];
  }
  if (i < cH * 2 * cH) W1h[i] = (_Float16)w1[i];
  if (i < cG) {
    biasC[i]      = b_ih_f[i] + b_hh_f[i];
    biasC[cG + i] = b_ih_b[i] + b_hh_b[i];
  }
}

// ---------------- Kernel 1: embedding gather + shared input projection ----------------
// P[b, n, p] = emb_pad[b,p] @ Wih[n,:] + biasC[n]   (f16, gate-major layout)
__global__ __launch_bounds__(256) void k_embed_proj(
    const int* __restrict__ input, const float* __restrict__ emb,
    const _Float16* __restrict__ Wih, const float* __restrict__ biasC,
    _Float16* __restrict__ P) {
  __shared__ _Float16 Aemb[16 * 128];
  const int b   = blockIdx.x / 9;
  const int p0  = (blockIdx.x % 9) * 16;
  const int tid = threadIdx.x;

  { // gather 16 embedding rows (f32 -> f16) into LDS, one b128 store each
    int r  = tid >> 4;
    int k0 = (tid & 15) * 8;
    int p  = p0 + r;
    int tok = -1;
    if (p >= cW && p < cS + cW) tok = input[b * cS + (p - cW)];
    union { uint4 q; _Float16 h[8]; } u;
    if (tok >= 0) {
      const float* src = emb + (size_t)tok * cE + k0;
      v4f f0 = *reinterpret_cast<const v4f*>(src);
      v4f f1 = *reinterpret_cast<const v4f*>(src + 4);
#pragma unroll
      for (int q = 0; q < 4; ++q) { u.h[q] = (_Float16)f0[q]; u.h[4 + q] = (_Float16)f1[q]; }
    } else {
      u.q = make_uint4(0u, 0u, 0u, 0u);
    }
    *reinterpret_cast<uint4*>(&Aemb[r * 128 + k0]) = u.q;
  }
  __syncthreads();

  const int wave = tid >> 5;
  const int lane = tid & 31;
  const int lm   = lane & 15;
  const int hb16 = (lane >> 4) & 1;

  v16h afr[4];
#pragma unroll
  for (int kk = 0; kk < 4; ++kk) afr[kk] = load_a_lds(Aemb, lane, kk);

#pragma unroll
  for (int j = 0; j < 8; ++j) {          // 8 waves x 8 tiles = 1024 gate cols
    const int n0 = (wave * 8 + j) * 16;
    const float bias = biasC[n0 + lm];
    v8f c;
#pragma unroll
    for (int r = 0; r < 8; ++r) c[r] = bias;
#pragma unroll
    for (int kk = 0; kk < 4; ++kk) {
      v16h bfr = load_b_h(Wih, n0, cE, lane, kk);
      c = wmma_f16(afr[kk], bfr, c);
    }
    // lane owns 8 consecutive positions p = p0 + hb16*8 + r at column n
    const int n = n0 + lm;
    union { uint4 q; _Float16 h[8]; } st;
#pragma unroll
    for (int r = 0; r < 8; ++r) st.h[r] = (_Float16)c[r];
    _Float16* dst = P + ((size_t)b * cNG + n) * cPSTR + p0 + hb16 * 8;
    *reinterpret_cast<uint4*>(dst) = st.q;
  }
}

// ---------------- Kernel 2: bidirectional LSTM + fused FC + score scatter ----------------
__global__ __launch_bounds__(256) void k_lstm_fc(
    const _Float16* __restrict__ P,
    const _Float16* __restrict__ Whh,   // [1024][128]: fwd rows 0..511, bwd 512..1023
    const _Float16* __restrict__ W1h,   // [128][256]
    const float* __restrict__ b1, const float* __restrict__ w2,
    const float* __restrict__ b2,
    float* __restrict__ col_sum) {
  __shared__ _Float16 hsave[cT][16 * 128];   // forward hidden states (f16)
  __shared__ _Float16 hcur[2][16 * 128];     // backward hidden state (double buffer)
  __shared__ float scores[16];

  const int b    = blockIdx.x >> 3;
  const int s0   = (blockIdx.x & 7) * 16;
  const int tid  = threadIdx.x;
  const int wave = tid >> 5;
  const int lane = tid & 31;
  const int lm   = lane & 15;
  const int hb16 = (lane >> 4) & 1;
  const float b2v = b2[0];
  const float b1v = b1[wave * 16 + lm];
  const float w2v = w2[wave * 16 + lm];

  // Recurrent weight fragments (register-resident across all timesteps).
  v16h Bhh[4][4];
#pragma unroll
  for (int j = 0; j < 4; ++j)
#pragma unroll
    for (int kk = 0; kk < 4; ++kk)
      Bhh[j][kk] = load_b_h(Whh, 128 * j + 16 * wave, cH, lane, kk);

  // Preload ALL gate pre-activations for this lane / direction:
  // lane's rows are p = s0 + hb16*8 + r, timestep t shifts by +t, so 16
  // consecutive halves cover r+t in [0,13]. Two b128 loads per gate tile.
  union PF { uint4 q[2]; _Float16 h[16]; };
  PF pf[4];
#pragma unroll
  for (int j = 0; j < 4; ++j) {
    const int n = 128 * j + 16 * wave + lm;
    const _Float16* src = P + ((size_t)b * cNG + n) * cPSTR + s0 + hb16 * 8;
    pf[j].q[0] = *reinterpret_cast<const uint4*>(src);
    pf[j].q[1] = *reinterpret_cast<const uint4*>(src + 8);
  }

  float cst[8];
#pragma unroll
  for (int r = 0; r < 8; ++r) cst[r] = 0.f;

  // ---- forward LSTM (fully unrolled: all indices compile-time) ----
#pragma unroll
  for (int t = 0; t < cT; ++t) {
    v16h afr[4];
    if (t > 0) {
#pragma unroll
      for (int kk = 0; kk < 4; ++kk) afr[kk] = load_a_lds(&hsave[t - 1][0], lane, kk);
    }
    v8f g[4];
#pragma unroll
    for (int j = 0; j < 4; ++j)
#pragma unroll
      for (int r = 0; r < 8; ++r)
        g[j][r] = (float)pf[j].h[r + t];
    if (t > 0) {
#pragma unroll
      for (int j = 0; j < 4; ++j)
#pragma unroll
        for (int kk = 0; kk < 4; ++kk)
          g[j] = wmma_f16(afr[kk], Bhh[j][kk], g[j]);
    }
#pragma unroll
    for (int r = 0; r < 8; ++r) {
      float iv = sigm(g[0][r]);
      float fv = sigm(g[1][r]);
      float gv = tanh_fast(g[2][r]);
      float ov = sigm(g[3][r]);
      cst[r] = fv * cst[r] + iv * gv;
      float hv = ov * tanh_fast(cst[r]);
      int m = r + hb16 * 8;
      hsave[t][m * 128 + wave * 16 + lm] = (_Float16)hv;
    }
    __syncthreads();
  }

  // swap to backward weights + FC1 weights + backward gate fragments
#pragma unroll
  for (int j = 0; j < 4; ++j)
#pragma unroll
    for (int kk = 0; kk < 4; ++kk)
      Bhh[j][kk] = load_b_h(Whh, cG + 128 * j + 16 * wave, cH, lane, kk);
  v16h Bfc[8];
#pragma unroll
  for (int kk = 0; kk < 8; ++kk)
    Bfc[kk] = load_b_h(W1h, 16 * wave, 2 * cH, lane, kk);
#pragma unroll
  for (int j = 0; j < 4; ++j) {
    const int n = cG + 128 * j + 16 * wave + lm;
    const _Float16* src = P + ((size_t)b * cNG + n) * cPSTR + s0 + hb16 * 8;
    pf[j].q[0] = *reinterpret_cast<const uint4*>(src);
    pf[j].q[1] = *reinterpret_cast<const uint4*>(src + 8);
  }

#pragma unroll
  for (int r = 0; r < 8; ++r) cst[r] = 0.f;

  // ---- backward LSTM + fused FC + score scatter (fully unrolled) ----
  // hcur double-buffered: read buf (t+1)&1, write buf t&1 -> only 2 barriers/step.
#pragma unroll
  for (int t = cT - 1; t >= 0; --t) {
    const int cur = t & 1, prv = (t + 1) & 1;
    v16h afr[4];
    if (t < cT - 1) {
#pragma unroll
      for (int kk = 0; kk < 4; ++kk) afr[kk] = load_a_lds(&hcur[prv][0], lane, kk);
    }
    v8f g[4];
#pragma unroll
    for (int j = 0; j < 4; ++j)
#pragma unroll
      for (int r = 0; r < 8; ++r)
        g[j][r] = (float)pf[j].h[r + t];
    if (t < cT - 1) {
#pragma unroll
      for (int j = 0; j < 4; ++j)
#pragma unroll
        for (int kk = 0; kk < 4; ++kk)
          g[j] = wmma_f16(afr[kk], Bhh[j][kk], g[j]);
    }
#pragma unroll
    for (int r = 0; r < 8; ++r) {
      float iv = sigm(g[0][r]);
      float fv = sigm(g[1][r]);
      float gv = tanh_fast(g[2][r]);
      float ov = sigm(g[3][r]);
      cst[r] = fv * cst[r] + iv * gv;
      float hv = ov * tanh_fast(cst[r]);
      int m = r + hb16 * 8;
      hcur[cur][m * 128 + wave * 16 + lm] = (_Float16)hv;
    }
    if (tid < 16) scores[tid] = b2v;   // re-init scores in same phase
    __syncthreads();                   // hcur[cur] + scores ready

    // FC1: [16 x 256] @ [256 x 128], wave w -> out cols 16w..16w+15
    v8f oc;
#pragma unroll
    for (int r = 0; r < 8; ++r) oc[r] = 0.f;
#pragma unroll
    for (int kk = 0; kk < 8; ++kk) {
      v16h af = load_a_fc(&hsave[t][0], &hcur[cur][0], lane, kk);
      oc = wmma_f16(af, Bfc[kk], oc);
    }
    // FC2: leaky(out1 + b1) . w2, reduce over the wave's 16 cols then over waves
#pragma unroll
    for (int r = 0; r < 8; ++r) {
      float v = oc[r] + b1v;
      v = v > 0.f ? v : 0.01f * v;
      v *= w2v;
      v += __shfl_xor(v, 1, 32);
      v += __shfl_xor(v, 2, 32);
      v += __shfl_xor(v, 4, 32);
      v += __shfl_xor(v, 8, 32);
      if (lm == 0) atomicAdd(&scores[r + hb16 * 8], v);
    }
    __syncthreads();                   // all score adds done
    if (tid < 16)
      atomicAdd(&col_sum[b * cPOS + s0 + tid + t], scores[tid]);
    // no barrier needed: only tid<16 touch scores next (program order),
    // and the next iteration's first barrier orders everything else.
  }
}

// ---------------- small epilogue kernels ----------------
__global__ __launch_bounds__(256) void k_zero(float* __restrict__ p, int n) {
  int i = blockIdx.x * 256 + threadIdx.x;
  if (i < n) p[i] = 0.f;
}

__global__ __launch_bounds__(256) void k_sigmoid(const float* __restrict__ col,
                                                 float* __restrict__ out) {
  int i = blockIdx.x * 256 + threadIdx.x;   // i < B*S
  int b = i >> 7;
  int s = i & 127;
  float v = col[b * cPOS + cW + s];
  out[i] = sigm(v);
}

// ---------------- host entry ----------------
extern "C" void kernel_launch(void* const* d_in, const int* in_sizes, int n_in,
                              void* d_out, int out_size, void* d_ws, size_t ws_size,
                              hipStream_t stream) {
  (void)in_sizes; (void)n_in; (void)out_size;
  const int*   input  = (const int*)d_in[0];
  const float* emb    = (const float*)d_in[1];
  const float* w_ih_f = (const float*)d_in[2];
  const float* w_hh_f = (const float*)d_in[3];
  const float* b_ih_f = (const float*)d_in[4];
  const float* b_hh_f = (const float*)d_in[5];
  const float* w_ih_b = (const float*)d_in[6];
  const float* w_hh_b = (const float*)d_in[7];
  const float* b_ih_b = (const float*)d_in[8];
  const float* b_hh_b = (const float*)d_in[9];
  const float* w1     = (const float*)d_in[10];
  const float* b1     = (const float*)d_in[11];
  const float* w2     = (const float*)d_in[12];
  const float* b2     = (const float*)d_in[13];
  float* out = (float*)d_out;

  // workspace layout (all 256B aligned)
  const size_t pBytes   = (size_t)cB * cPSTR * cNG * sizeof(_Float16); // ~75.5 MB
  const size_t colBytes = (size_t)cB * cPOS * sizeof(float);           // 137 KB
  const size_t wihBytes = (size_t)cNG * cE * sizeof(_Float16);         // 256 KB
  const size_t whhBytes = (size_t)cNG * cH * sizeof(_Float16);         // 256 KB
  const size_t w1Bytes  = (size_t)cH * 2 * cH * sizeof(_Float16);      // 64 KB
  const size_t biasBytes = (size_t)cNG * sizeof(float);                // 4 KB
  if (ws_size < pBytes + colBytes + wihBytes + whhBytes + w1Bytes + biasBytes) return;

  char* wp = (char*)d_ws;
  _Float16* P    = (_Float16*)wp;                 wp += pBytes;
  float*    col  = (float*)wp;                    wp += colBytes;
  _Float16* Wih  = (_Float16*)wp;                 wp += wihBytes;
  _Float16* Whh  = (_Float16*)wp;                 wp += whhBytes;
  _Float16* W1h  = (_Float16*)wp;                 wp += w1Bytes;
  float*    biasC = (float*)wp;

  k_cvt<<<(cG * cE) / 256, 256, 0, stream>>>(w_ih_f, w_ih_b, w_hh_f, w_hh_b, w1,
                                             b_ih_f, b_hh_f, b_ih_b, b_hh_b,
                                             Wih, Whh, W1h, biasC);
  const int nCol = cB * cPOS;
  k_zero<<<(nCol + 255) / 256, 256, 0, stream>>>(col, nCol);
  k_embed_proj<<<cB * 9, 256, 0, stream>>>(input, emb, Wih, biasC, P);
  k_lstm_fc<<<cB * 8, 256, 0, stream>>>(P, Whh, W1h, b1, w2, b2, col);
  k_sigmoid<<<(cB * cS) / 256, 256, 0, stream>>>(col, out);
}
